// Mamba3Block_58841051955393
// MI455X (gfx1250) — matrix operand order repaired and checked by reading, not compiled, not run on hardware
//
#include <hip/hip_runtime.h>

#define B_SZ 8
#define SEQ  2048
#define DM   1024            // D_MODEL == D_STATE
#define NWG  32              // workgroups in persistent recurrence kernel
#define COLS 32              // columns of A / Wc owned per recurrence WG

typedef __attribute__((ext_vector_type(16))) __bf16 v16bf;
typedef __attribute__((ext_vector_type(8)))  float  v8f;

union BFrag { v16bf v; uint4 q[2]; unsigned short us[16]; };
union Acc   { v8f  v; float f[8]; };

__device__ inline unsigned short f2bf(float f) {
    unsigned u = __builtin_bit_cast(unsigned, f);
    unsigned r = u + 0x7FFFu + ((u >> 16) & 1u);   // round-to-nearest-even
    return (unsigned short)(r >> 16);
}

// ---------------------------------------------------------------- weights->bf16
__global__ __launch_bounds__(256)
void k_convert_weights(const float* __restrict__ A, const float* __restrict__ Wb,
                       const float* __restrict__ Wc,
                       unsigned short* __restrict__ Abf,
                       unsigned short* __restrict__ Wbbf,
                       unsigned short* __restrict__ Wcbf) {
    int i = blockIdx.x * 256 + threadIdx.x;
    if (i < DM * DM) {
        Abf[i]  = f2bf(A[i]);
        Wbbf[i] = f2bf(Wb[i]);
        Wcbf[i] = f2bf(Wc[i]);
    }
}

__global__ void k_init_sync(unsigned* sync) {
    if (threadIdx.x == 0) { sync[0] = 0u; sync[1] = 0u; }
}

// ---------------------------------------------------------------- LayerNorm
__global__ __launch_bounds__(256)
void k_layernorm(const float* __restrict__ x, const float* __restrict__ g,
                 const float* __restrict__ b, float* __restrict__ xn) {
    __shared__ float red[256];
    int tid = threadIdx.x;
    size_t base = (size_t)blockIdx.x * DM;
    float v[4];
    float s = 0.f;
#pragma unroll
    for (int k = 0; k < 4; k++) { v[k] = x[base + tid + k * 256]; s += v[k]; }
    red[tid] = s; __syncthreads();
    for (int off = 128; off > 0; off >>= 1) {
        if (tid < off) red[tid] += red[tid + off];
        __syncthreads();
    }
    float mu = red[0] * (1.f / DM);
    __syncthreads();
    float s2 = 0.f;
#pragma unroll
    for (int k = 0; k < 4; k++) { float d = v[k] - mu; s2 += d * d; }
    red[tid] = s2; __syncthreads();
    for (int off = 128; off > 0; off >>= 1) {
        if (tid < off) red[tid] += red[tid + off];
        __syncthreads();
    }
    float var = red[0] * (1.f / DM);
    float rs  = rsqrtf(var + 1e-5f);
#pragma unroll
    for (int k = 0; k < 4; k++) {
        int c = tid + k * 256;
        xn[base + c] = (v[k] - mu) * rs * g[c] + b[c];
    }
}

// ---------------------------------------------------------------- depthwise conv
// xc[b,t,c] = cb[c] + sum_{j=0..3} cw[c,j] * xn[b, t+j-2, c]; also emit bf16 copy
__global__ __launch_bounds__(256)
void k_conv(const float* __restrict__ xn, const float* __restrict__ cw,
            const float* __restrict__ cb, float* __restrict__ xc,
            unsigned short* __restrict__ xcbf) {
    int t = blockIdx.x & (SEQ - 1);
    int b = blockIdx.x >> 11;                 // SEQ = 2^11
#pragma unroll
    for (int k = 0; k < 4; k++) {
        int c = threadIdx.x + k * 256;
        float4 w = *(const float4*)(cw + (size_t)c * 4);
        const float* wp = (const float*)&w;
        float acc = cb[c];
#pragma unroll
        for (int j = 0; j < 4; j++) {
            int tt = t + j - 2;
            if (tt >= 0 && tt < SEQ)
                acc += wp[j] * xn[(((size_t)b * SEQ) + tt) * DM + c];
        }
        size_t idx = (((size_t)b * SEQ) + t) * DM + c;
        xc[idx]   = acc;
        xcbf[idx] = f2bf(acc);
    }
}

// ---------------------------------------------------------------- pre = xc @ Wb^T
// pre[r,j] = sum_i xc[r,i] * Wb[j,i].  M=16384, N=1024, K=1024, bf16 WMMA, f32 acc.
// Each wave computes a 16x64 strip: one A fragment reused by 4 accumulators.
__global__ __launch_bounds__(256)
void k_pre_gemm(const unsigned short* __restrict__ xcbf,
                const unsigned short* __restrict__ Wbbf,
                float* __restrict__ pre) {
    int wave = threadIdx.x >> 5, lane = threadIdx.x & 31;
    int l = lane & 15, hi = lane >> 4;
    int id = blockIdx.x * 8 + wave;           // 0 .. 16383
    int tile_m = id >> 4;                     // 0 .. 1023
    int n_base = (id & 15) * 64;              // 0 .. 960

    const unsigned short* rowp = xcbf + (size_t)(tile_m * 16 + l) * DM;
    Acc acc[4];
#pragma unroll
    for (int s = 0; s < 4; s++)
#pragma unroll
        for (int r = 0; r < 8; r++) acc[s].f[r] = 0.f;

    for (int k0 = 0; k0 < DM; k0 += 32) {
        // A fragment: row M=l; us[0..7]=K(k0+hi*8..), us[8..15]=K(k0+16+hi*8..)
        BFrag a;
        const unsigned short* ap = rowp + k0 + hi * 8;
        a.q[0] = *(const uint4*)ap;
        a.q[1] = *(const uint4*)(ap + 16);
        // B fragments: col N=l; us[0..7]=K(k0+hi*8..), us[8..15]=K(k0+16+hi*8..)
        BFrag bb[4];
#pragma unroll
        for (int s = 0; s < 4; s++) {
            const unsigned short* wr =
                Wbbf + (size_t)(n_base + s * 16 + l) * DM + k0 + hi * 8;
            bb[s].q[0] = *(const uint4*)wr;
            bb[s].q[1] = *(const uint4*)(wr + 16);
        }
#pragma unroll
        for (int s = 0; s < 4; s++)
            acc[s].v = __builtin_amdgcn_wmma_f32_16x16x32_bf16(
                false, a.v, false, bb[s].v, (short)0, acc[s].v, false, false);
    }
#pragma unroll
    for (int s = 0; s < 4; s++)
#pragma unroll
        for (int r = 0; r < 8; r++) {
            int row = tile_m * 16 + r + hi * 8;   // C layout: lanes16-31 hold M=8..15
            int col = n_base + s * 16 + l;
            pre[(size_t)row * DM + col] = acc[s].f[r];
        }
}

// ---------------------------------------------------------------- recurrence
// Persistent kernel, NWG=32 WGs, each owns COLS=32 columns of A and Wc in LDS.
// Per step t: waves 0-1 compute h_t slice = tanh(h_{t-1}@A + pre_t) -> global,
// release fence, grid barrier, acquire fence, async-copy full h_t into LDS,
// waves 2-3 compute y_t slice = h_t@Wc_slice + D*xc + x.
__global__ __launch_bounds__(128)
void k_recurrence(const float* __restrict__ x, const float* __restrict__ xc,
                  const float* __restrict__ pre,
                  const unsigned short* __restrict__ Abf,
                  const unsigned short* __restrict__ Wcbf,
                  const float* __restrict__ Dv,
                  unsigned short* __restrict__ Hg, unsigned* __restrict__ sync,
                  float* __restrict__ out) {
    extern __shared__ unsigned short smem[];
    unsigned short* As = smem;                 // [COLS][DM]  As[n*DM+k] = A[k, j0+n]
    unsigned short* Ws = As + COLS * DM;       // [COLS][DM]  Ws[n*DM+k] = Wc[j0+n, k]
    unsigned short* hb = Ws + COLS * DM;       // [16][DM]    padded batch state (bf16)

    int tid = threadIdx.x, wave = tid >> 5, lane = tid & 31;
    int l = lane & 15, hi = lane >> 4;
    int j0 = blockIdx.x * COLS;

    for (int i = tid; i < COLS * DM; i += 128) {
        int n = i >> 10, k = i & (DM - 1);
        As[i] = Abf[(size_t)k * DM + j0 + n];          // transpose A column slice
    }
    for (int i = tid; i < COLS * DM; i += 128) {
        int n = i >> 10, k = i & (DM - 1);
        Ws[i] = Wcbf[(size_t)(j0 + n) * DM + k];       // Wc rows are contiguous
    }
    for (int i = tid; i < 16 * DM; i += 128) hb[i] = 0;   // h_{-1} = 0, pad rows 8..15 = 0
    __syncthreads();

    for (int t = 0; t < SEQ; t++) {
        unsigned short* Hcur = Hg + (size_t)(t & 1) * 16 * DM;

        if (wave < 2) {                                // ---- h update tiles
            int n0  = wave * 16;
            int col = j0 + n0 + l;
            Acc acc;
#pragma unroll
            for (int r = 0; r < 8; r++) {
                int b = r + hi * 8;
                acc.f[r] = (b < B_SZ) ? pre[((size_t)b * SEQ + t) * DM + col] : 0.f;
            }
            for (int k0 = 0; k0 < DM; k0 += 32) {
                BFrag a, bb;
                const unsigned short* ap = hb + l * DM + k0 + hi * 8;
                a.q[0] = *(const uint4*)ap;
                a.q[1] = *(const uint4*)(ap + 16);
                const unsigned short* bp = As + (n0 + l) * DM + k0 + hi * 8;
                bb.q[0] = *(const uint4*)bp;
                bb.q[1] = *(const uint4*)(bp + 16);
                acc.v = __builtin_amdgcn_wmma_f32_16x16x32_bf16(
                    false, a.v, false, bb.v, (short)0, acc.v, false, false);
            }
#pragma unroll
            for (int r = 0; r < 8; r++) {
                int b = r + hi * 8;                    // pad rows compute tanh(0)=0
                Hcur[b * DM + col] = f2bf(tanhf(acc.f[r]));
            }
            if (t + 1 < SEQ && hi == 0) {              // warm next step's pre (L2)
#pragma unroll
                for (int r = 0; r < 8; r++)
                    __builtin_prefetch(&pre[((size_t)r * SEQ + t + 1) * DM + col], 0, 1);
            }
        } else {                                       // ---- y waves: warm xc/x lines
            int n0  = (wave - 2) * 16;
            int col = j0 + n0 + l;
            if (hi == 0) {
#pragma unroll
                for (int r = 0; r < 8; r++) {
                    size_t idx = ((size_t)r * SEQ + t) * DM + col;
                    __builtin_prefetch(&xc[idx], 0, 1);
                    __builtin_prefetch(&x[idx], 0, 1);
                }
            }
        }

        // ---- device-wide barrier: release, arrive/spin, acquire
        __threadfence();                               // all waves drain h stores
        __syncthreads();
        if (tid == 0) {
            unsigned g = atomicAdd(&sync[1], 0u);
            if (atomicAdd(&sync[0], 1u) == (unsigned)(NWG - 1)) {
                atomicExch(&sync[0], 0u);
                __threadfence();
                atomicAdd(&sync[1], 1u);
            } else {
                while (atomicAdd(&sync[1], 0u) == g) __builtin_amdgcn_s_sleep(4);
            }
        }
        __syncthreads();
        __threadfence();                               // acquire: fresh view of Hcur

        // ---- async copy h_t (32 KB) global -> LDS, ASYNCcnt-tracked
        {
            // 16*DM ushorts = 2048 16-byte chunks, 16 per thread
            for (int i = tid; i < (16 * DM) / 8; i += 128) {
                unsigned ldsAddr = (unsigned)(size_t)(hb + (size_t)i * 8);
                const unsigned short* gp = Hcur + (size_t)i * 8;
                asm volatile("global_load_async_to_lds_b128 %0, %1, off"
                             :: "v"(ldsAddr), "v"(gp) : "memory");
            }
            asm volatile("s_wait_asynccnt 0x0" ::: "memory");
        }
        __syncthreads();

        if (wave >= 2) {                               // ---- y tiles (use h_t)
            int n0  = (wave - 2) * 16;
            int col = j0 + n0 + l;
            Acc acc;
#pragma unroll
            for (int r = 0; r < 8; r++) acc.f[r] = 0.f;
            for (int k0 = 0; k0 < DM; k0 += 32) {
                BFrag a, bb;
                const unsigned short* ap = hb + l * DM + k0 + hi * 8;
                a.q[0] = *(const uint4*)ap;
                a.q[1] = *(const uint4*)(ap + 16);
                const unsigned short* bp = Ws + (n0 + l) * DM + k0 + hi * 8;
                bb.q[0] = *(const uint4*)bp;
                bb.q[1] = *(const uint4*)(bp + 16);
                acc.v = __builtin_amdgcn_wmma_f32_16x16x32_bf16(
                    false, a.v, false, bb.v, (short)0, acc.v, false, false);
            }
#pragma unroll
            for (int r = 0; r < 8; r++) {
                int b = r + hi * 8;
                if (b < B_SZ) {
                    size_t idx = ((size_t)b * SEQ + t) * DM + col;
                    out[idx] = acc.f[r] + Dv[col] * xc[idx] + x[idx];
                }
            }
        }
    }
}

// ---------------------------------------------------------------- launcher
extern "C" void kernel_launch(void* const* d_in, const int* in_sizes, int n_in,
                              void* d_out, int out_size, void* d_ws, size_t ws_size,
                              hipStream_t stream) {
    (void)in_sizes; (void)n_in; (void)out_size; (void)ws_size;
    const float* x  = (const float*)d_in[0];
    const float* A  = (const float*)d_in[1];
    const float* Wb = (const float*)d_in[2];
    const float* Wc = (const float*)d_in[3];
    const float* Dv = (const float*)d_in[4];
    const float* cw = (const float*)d_in[5];
    const float* cb = (const float*)d_in[6];
    const float* lg = (const float*)d_in[7];
    const float* lb = (const float*)d_in[8];
    float* out = (float*)d_out;

    char* ws = (char*)d_ws;
    size_t o = 0;
    float* xc  = (float*)(ws + o); o += (size_t)B_SZ * SEQ * DM * 4;            // 64 MB
    unsigned short* xcbf = (unsigned short*)(ws + o); o += (size_t)B_SZ * SEQ * DM * 2; // 32 MB
    float* pre = (float*)(ws + o); o += (size_t)B_SZ * SEQ * DM * 4;            // 64 MB
    unsigned short* Abf  = (unsigned short*)(ws + o); o += (size_t)DM * DM * 2;
    unsigned short* Wbbf = (unsigned short*)(ws + o); o += (size_t)DM * DM * 2;
    unsigned short* Wcbf = (unsigned short*)(ws + o); o += (size_t)DM * DM * 2;
    unsigned short* Hg   = (unsigned short*)(ws + o); o += (size_t)2 * 16 * DM * 2;
    unsigned* sync = (unsigned*)(ws + o); o += 256;

    k_convert_weights<<<DM * DM / 256, 256, 0, stream>>>(A, Wb, Wc, Abf, Wbbf, Wcbf);
    k_init_sync<<<1, 32, 0, stream>>>(sync);
    k_layernorm<<<B_SZ * SEQ, 256, 0, stream>>>(x, lg, lb, out);   // d_out as xn scratch
    k_conv<<<B_SZ * SEQ, 256, 0, stream>>>(out, cw, cb, xc, xcbf);
    k_pre_gemm<<<2048, 256, 0, stream>>>(xcbf, Wbbf, pre);         // 16384 wave-tiles / 8
    size_t lds = (size_t)(2 * COLS * DM + 16 * DM) * sizeof(unsigned short); // 160 KB
    k_recurrence<<<NWG, 128, lds, stream>>>(x, xc, pre, Abf, Wcbf, Dv, Hg, sync, out);
}
